// HRMHead_21912923144843
// MI455X (gfx1250) — matrix operand order repaired and checked by reading, not compiled, not run on hardware
//
#include <hip/hip_runtime.h>
#include <hip/hip_bf16.h>

typedef _Float16 half_t;
typedef __attribute__((ext_vector_type(8)))  _Float16 v8h;
typedef __attribute__((ext_vector_type(16))) _Float16 v16h;
typedef __attribute__((ext_vector_type(8)))  float    v8f;

#define B_TOT 1024
#define T_LEN 512
#define F_IN  128
#define HIGH_PERIOD 10
#define HEAD_HID 64

// ---------------- LDS layout (units: halves, f16) ----------------
// Row strides padded by +8 halves (16B) so 16 lanes reading 16B chunks at
// stride*lane hit distinct bank groups (stride/16 is odd -> coprime with 16).
#define XCAT_S 200   // [x(128) | h_high(64)] rows, K=192 used
#define HLOW_S 72    // h_low rows, K=64 used
#define WLI_S  200   // low_Wih  rows (K=192)
#define WLH_S  72    // low_Whh  rows (K=64)
#define WHI_S  136   // high_Wih rows (K=128)
#define WHH_S  72    // high_Whh rows (K=64)

#define XCAT_OFF 0
#define HLOW_OFF (XCAT_OFF + 16 * XCAT_S)      // 3200
#define WLI_OFF  (HLOW_OFF + 16 * HLOW_S)      // 4352
#define WLH_OFF  (WLI_OFF + 192 * WLI_S)       // 42752
#define WHI_OFF  (WLH_OFF + 192 * WLH_S)       // 56576
#define WHH_OFF  (WHI_OFF + 192 * WHI_S)       // 82688
#define HALVES_TOTAL (WHH_OFF + 192 * WHH_S)   // 96512 halves = 193024 B

// fp32 region appended after the f16 region (units: floats)
#define FREG_OFF_BYTES (HALVES_TOTAL * 2)
#define BI_LI 0
#define BI_LH 192
#define BI_HI 384
#define BI_HH 576
#define HFIN  768                                 // 16x64 fp32 final h_low
#define FLOATS_TOTAL (HFIN + 16 * 64)             // 1792
#define SMEM_BYTES (FREG_OFF_BYTES + FLOATS_TOTAL * 4)   // 200192 B < 320KB WGP LDS

// ---------------- helpers ----------------
__device__ __forceinline__ v8f wmma_f16(v16h a, v16h b, v8f c) {
  // D = A(16x32,f16) x B(32x16,f16) + C(16x16,f32)
  return __builtin_amdgcn_wmma_f32_16x16x32_f16(
      /*neg_a=*/false, a, /*neg_b=*/false, b,
      /*c_mod=*/(short)0, c, /*reuse_a=*/false, /*reuse_b=*/false);
}

// A-matrix fragment (16x32 f16): lane holds row m=lane&15, K halves at
// {koff..koff+7} and {koff+16..koff+23}, koff already includes 8*(lane>=16).
__device__ __forceinline__ v16h ldA(const half_t* p) {
  v8h lo = *(const v8h*)(p);
  v8h hi = *(const v8h*)(p + 16);
  return __builtin_shufflevector(lo, hi, 0,1,2,3,4,5,6,7,8,9,10,11,12,13,14,15);
}

// B-matrix fragment (32x16 f16): lane holds column n=lane&15, K halves at
// {koff..koff+7} and {koff+16..koff+23}, koff already includes 8*(lane>=16).
__device__ __forceinline__ v16h ldB(const half_t* p) {
  v8h lo = *(const v8h*)(p);
  v8h hi = *(const v8h*)(p + 16);
  return __builtin_shufflevector(lo, hi, 0,1,2,3,4,5,6,7,8,9,10,11,12,13,14,15);
}

__device__ __forceinline__ float sigm(float x) {
  return 1.0f / (1.0f + __expf(-x));
}
__device__ __forceinline__ float tanh_fast(float x) {
  x = fminf(15.0f, fmaxf(-15.0f, x));
  float e = __expf(2.0f * x);
  return (e - 1.0f) / (e + 1.0f);
}

// ---------------- kernel ----------------
__global__ void __launch_bounds__(128, 1)
hrm_hier_gru_kernel(const float* __restrict__ x,
                    const float* __restrict__ hWih, const float* __restrict__ hWhh,
                    const float* __restrict__ hbih, const float* __restrict__ hbhh,
                    const float* __restrict__ lWih, const float* __restrict__ lWhh,
                    const float* __restrict__ lbih, const float* __restrict__ lbhh,
                    const float* __restrict__ W1, const float* __restrict__ b1,
                    const float* __restrict__ W2, const float* __restrict__ b2,
                    float* __restrict__ out) {
  extern __shared__ char smem[];
  half_t* lh = (half_t*)smem;
  float*  lf = (float*)(smem + FREG_OFF_BYTES);

  const int tid = threadIdx.x;
  const int b0  = blockIdx.x * 16;   // 16 batch rows per block

  // ---- one-time staging: weights fp32 -> f16 LDS, biases fp32 LDS, h=0 ----
  for (int idx = tid; idx < 192 * 192; idx += 128) {
    int r = idx / 192, c = idx - r * 192;
    lh[WLI_OFF + r * WLI_S + c] = (half_t)lWih[idx];
  }
  for (int idx = tid; idx < 192 * 64; idx += 128) {
    int r = idx >> 6, c = idx & 63;
    lh[WLH_OFF + r * WLH_S + c] = (half_t)lWhh[idx];
  }
  for (int idx = tid; idx < 192 * 128; idx += 128) {
    int r = idx >> 7, c = idx & 127;
    lh[WHI_OFF + r * WHI_S + c] = (half_t)hWih[idx];
  }
  for (int idx = tid; idx < 192 * 64; idx += 128) {
    int r = idx >> 6, c = idx & 63;
    lh[WHH_OFF + r * WHH_S + c] = (half_t)hWhh[idx];
  }
  for (int i = tid; i < 192; i += 128) {
    lf[BI_LI + i] = lbih[i];
    lf[BI_LH + i] = lbhh[i];
    lf[BI_HI + i] = hbih[i];
    lf[BI_HH + i] = hbhh[i];
  }
  for (int i = tid; i < 16 * 64; i += 128) {       // h_high = 0 (xcat cols 128..191)
    int r = i >> 6, c = i & 63;
    lh[XCAT_OFF + r * XCAT_S + 128 + c] = (half_t)0.0f;
  }
  for (int i = tid; i < 16 * HLOW_S; i += 128)     // h_low = 0
    lh[HLOW_OFF + i] = (half_t)0.0f;
  __syncthreads();

  // ---- per-lane loop-invariant state ----
  const int wave  = tid >> 5;         // 0..3 -> owns hidden cols 16*wave..+15
  const int lane  = tid & 31;
  const int nlo   = lane & 15;
  const int hi16  = lane >> 4;
  const int kA    = hi16 * 8;         // A-frag K sub-offset
  const int kB    = hi16 * 8;         // B-frag K sub-offset
  const int hc    = wave * 16 + nlo;  // hidden column 0..63
  const int rbase = hi16 * 8;         // C-layout row base

  const float lbir = lf[BI_LI + hc], lbiz = lf[BI_LI + 64 + hc], lbin = lf[BI_LI + 128 + hc];
  const float lbhr = lf[BI_LH + hc], lbhz = lf[BI_LH + 64 + hc], lbhn = lf[BI_LH + 128 + hc];
  const float gbir = lf[BI_HI + hc], gbiz = lf[BI_HI + 64 + hc], gbin = lf[BI_HI + 128 + hc];
  const float gbhr = lf[BI_HH + hc], gbhz = lf[BI_HH + 64 + hc], gbhn = lf[BI_HH + 128 + hc];

  const half_t* pAx = lh + XCAT_OFF + nlo * XCAT_S + kA;   // A rows: [x | h_high]
  const half_t* pAl = lh + HLOW_OFF + nlo * HLOW_S + kA;   // A rows: h_low
  const half_t* pLIr = lh + WLI_OFF + (hc      ) * WLI_S + kB;
  const half_t* pLIz = lh + WLI_OFF + ( 64 + hc) * WLI_S + kB;
  const half_t* pLIn = lh + WLI_OFF + (128 + hc) * WLI_S + kB;
  const half_t* pLHr = lh + WLH_OFF + (hc      ) * WLH_S + kB;
  const half_t* pLHz = lh + WLH_OFF + ( 64 + hc) * WLH_S + kB;
  const half_t* pLHn = lh + WLH_OFF + (128 + hc) * WLH_S + kB;
  const half_t* pHIr = lh + WHI_OFF + (hc      ) * WHI_S + kB;
  const half_t* pHIz = lh + WHI_OFF + ( 64 + hc) * WHI_S + kB;
  const half_t* pHIn = lh + WHI_OFF + (128 + hc) * WHI_S + kB;
  const half_t* pHHr = lh + WHH_OFF + (hc      ) * WHH_S + kB;
  const half_t* pHHz = lh + WHH_OFF + ( 64 + hc) * WHH_S + kB;
  const half_t* pHHn = lh + WHH_OFF + (128 + hc) * WHH_S + kB;

  half_t* hlow_w  = lh + HLOW_OFF + hc;          // + row*HLOW_S
  half_t* hhigh_w = lh + XCAT_OFF + 128 + hc;    // + row*XCAT_S

  v8f hlowf  = {};   // fp32 h_low tile, persistent in registers
  v8f hhighf = {};   // fp32 h_high tile

  // x staging assignment: thread -> (row, 16-col chunk)
  const int xr = tid >> 3;
  const int xf = (tid & 7) << 4;
  const float* xbase = x + (size_t)(b0 + xr) * T_LEN * F_IN + xf;
  half_t* xdst = lh + XCAT_OFF + xr * XCAT_S + xf;

  for (int t = 0; t < T_LEN; ++t) {
    // ---- stage x_t tile: fp32 -> f16 LDS ----
    {
      const float4* xs = (const float4*)(xbase + (size_t)t * F_IN);
      float4 u0 = xs[0], u1 = xs[1], u2 = xs[2], u3 = xs[3];
      v8h d0 = {(half_t)u0.x, (half_t)u0.y, (half_t)u0.z, (half_t)u0.w,
                (half_t)u1.x, (half_t)u1.y, (half_t)u1.z, (half_t)u1.w};
      v8h d1 = {(half_t)u2.x, (half_t)u2.y, (half_t)u2.z, (half_t)u2.w,
                (half_t)u3.x, (half_t)u3.y, (half_t)u3.z, (half_t)u3.w};
      *(v8h*)(xdst)     = d0;
      *(v8h*)(xdst + 8) = d1;
    }
    __syncthreads();  // x visible; prev-step h_low writes visible

    // ---- high GRU (uniform branch; only every 10th step) ----
    if ((t % HIGH_PERIOD) == 0) {
      v8f ar = {}, az = {}, an = {}, br = {}, bz = {}, bn = {};
#pragma unroll
      for (int kb = 0; kb < 4; ++kb) {              // gi: K=128 over x
        v16h a = ldA(pAx + 32 * kb);
        ar = wmma_f16(a, ldB(pHIr + 32 * kb), ar);
        az = wmma_f16(a, ldB(pHIz + 32 * kb), az);
        an = wmma_f16(a, ldB(pHIn + 32 * kb), an);
      }
#pragma unroll
      for (int kb = 0; kb < 2; ++kb) {              // gh: K=64 over h_high
        v16h a = ldA(pAx + 128 + 32 * kb);
        br = wmma_f16(a, ldB(pHHr + 32 * kb), br);
        bz = wmma_f16(a, ldB(pHHz + 32 * kb), bz);
        bn = wmma_f16(a, ldB(pHHn + 32 * kb), bn);
      }
      __syncthreads();  // everyone done reading old h_high
#pragma unroll
      for (int v = 0; v < 8; ++v) {
        float r = sigm(ar[v] + gbir + br[v] + gbhr);
        float z = sigm(az[v] + gbiz + bz[v] + gbhz);
        float n = tanh_fast(an[v] + gbin + r * (bn[v] + gbhn));
        float h = (1.0f - z) * n + z * hhighf[v];
        hhighf[v] = h;
        hhigh_w[(v + rbase) * XCAT_S] = (half_t)h;
      }
      __syncthreads();  // new h_high visible for low GRU
    }

    // ---- low GRU (every step) ----
    {
      v8f ar = {}, az = {}, an = {}, br = {}, bz = {}, bn = {};
#pragma unroll
      for (int kb = 0; kb < 2; ++kb) {              // gh: K=64 over old h_low
        v16h a = ldA(pAl + 32 * kb);
        br = wmma_f16(a, ldB(pLHr + 32 * kb), br);
        bz = wmma_f16(a, ldB(pLHz + 32 * kb), bz);
        bn = wmma_f16(a, ldB(pLHn + 32 * kb), bn);
      }
#pragma unroll
      for (int kb = 0; kb < 6; ++kb) {              // gi: K=192 over [x | h_high]
        v16h a = ldA(pAx + 32 * kb);
        ar = wmma_f16(a, ldB(pLIr + 32 * kb), ar);
        az = wmma_f16(a, ldB(pLIz + 32 * kb), az);
        an = wmma_f16(a, ldB(pLIn + 32 * kb), an);
      }
      __syncthreads();  // everyone done reading old h_low / xcat
#pragma unroll
      for (int v = 0; v < 8; ++v) {
        float r = sigm(ar[v] + lbir + br[v] + lbhr);
        float z = sigm(az[v] + lbiz + bz[v] + lbhz);
        float n = tanh_fast(an[v] + lbin + r * (bn[v] + lbhn));
        float h = (1.0f - z) * n + z * hlowf[v];
        hlowf[v] = h;
        hlow_w[(v + rbase) * HLOW_S] = (half_t)h;  // f16 copy for next matmul
      }
      // writes ordered before next iteration's __syncthreads
    }
  }

  // ---- head: relu(h @ W1^T + b1) @ W2^T + b2, sigmoid (tiny; scalar fp32) ----
#pragma unroll
  for (int v = 0; v < 8; ++v)
    lf[HFIN + (v + rbase) * 64 + hc] = hlowf[v];
  __syncthreads();

  if (tid < 16) {
    const float* hrow = lf + HFIN + tid * 64;
    float p = b2[0];
    for (int i = 0; i < HEAD_HID; ++i) {
      const float* w = W1 + i * 64;
      float acc = b1[i];
      for (int k = 0; k < 64; ++k) acc = fmaf(hrow[k], w[k], acc);
      acc = fmaxf(acc, 0.0f);
      p = fmaf(W2[i], acc, p);
    }
    out[b0 + tid] = 1.0f / (1.0f + __expf(-p));
  }
}

// ---------------- host launch ----------------
extern "C" void kernel_launch(void* const* d_in, const int* in_sizes, int n_in,
                              void* d_out, int out_size, void* d_ws, size_t ws_size,
                              hipStream_t stream) {
  (void)in_sizes; (void)n_in; (void)out_size; (void)d_ws; (void)ws_size;
  const float* x    = (const float*)d_in[0];
  const float* hWih = (const float*)d_in[1];
  const float* hWhh = (const float*)d_in[2];
  const float* hbih = (const float*)d_in[3];
  const float* hbhh = (const float*)d_in[4];
  const float* lWih = (const float*)d_in[5];
  const float* lWhh = (const float*)d_in[6];
  const float* lbih = (const float*)d_in[7];
  const float* lbhh = (const float*)d_in[8];
  const float* W1   = (const float*)d_in[9];
  const float* b1   = (const float*)d_in[10];
  const float* W2   = (const float*)d_in[11];
  const float* b2   = (const float*)d_in[12];
  float* out = (float*)d_out;

  hrm_hier_gru_kernel<<<dim3(B_TOT / 16), dim3(128), SMEM_BYTES, stream>>>(
      x, hWih, hWhh, hbih, hbhh, lWih, lWhh, lbih, lbhh, W1, b1, W2, b2, out);
}
